// Decoder_38491496907198
// MI455X (gfx1250) — matrix operand order repaired and checked by reading, not compiled, not run on hardware
//
#include <hip/hip_runtime.h>
#include <math.h>

// ---------------------------------------------------------------------------
// Tacotron2 decoder for MI455X (gfx1250, wave32, WMMA).
//
// Latency-bound sequential scan (T=400, ~600 MFLOP/step). Weights (~35MB f16)
// are L2-resident (192MB L2): per-step GEMMs are L2-latency bound, not HBM
// bound. B=16 == WMMA M-tile, so every GEMM is a row of
// v_wmma_f32_16x16x32_f16 tiles with fp32 accumulation, register
// double-buffered to overlap L2 loads with the matrix pipe. One persistent
// kernel; 9 atomic grid barriers per step; the whole attention chain
// (q/conv/energies/softmax/context) is block-local (LDS + __syncthreads).
// ---------------------------------------------------------------------------

#define NBLK    64
#define THREADS 256
#define WPB     (THREADS/32)       // 8 waves per block
#define NWAVES  (NBLK*WPB)         // 512 waves total

#define B_    16
#define S_    256
#define T_    400
#define NMEL  80
#define PRE_  256
#define ENC_  512
#define HID_  1024
#define ATT_  128
#define NF_   32
#define KC_   31

typedef _Float16 f16;
typedef __attribute__((ext_vector_type(16))) _Float16 v16h;
typedef __attribute__((ext_vector_type(2)))  __fp16   h2v;   // cvt_pkrtz result type
typedef __attribute__((ext_vector_type(8)))  float    v8f;

static __device__ __forceinline__ float sigm(float x){ return 1.f/(1.f+__expf(-x)); }

union AFrag { v16h v; f16 u[16]; h2v p[8]; };

static __device__ __forceinline__ v8f wmma16(v16h a, v16h b, v8f c){
  return __builtin_amdgcn_wmma_f32_16x16x32_f16(false, a, false, b, (short)0, c, false, false);
}

// Register-double-buffered K-loop: issue loads for tile kt+1 before the WMMA
// for tile kt so s_wait_loadcnt can tolerate outstanding loads.
template <int NKT>
static __device__ __forceinline__ v8f gemm_loop(const f16* __restrict__ ap,
                                                const f16* __restrict__ wp, v8f acc){
  v16h a0 = *(const v16h*)ap;
  v16h b0 = *(const v16h*)wp;
  #pragma unroll 2
  for (int kt = 1; kt < NKT; ++kt){
    ap += 512; wp += 512;
    v16h a1 = *(const v16h*)ap;
    v16h b1 = *(const v16h*)wp;
    __builtin_prefetch(wp + 512, 0, 0);
    acc = wmma16(a0, b0, acc);
    a0 = a1; b0 = b1;
  }
  return wmma16(a0, b0, acc);
}

// Build one A-fragment (16x32 f16) for lane `lane` of k-tile kt.
// A layout: lane l: m=l&15, hi=l>>4; slots 0..7 -> k = kt*32+hi*8+s,
// slots 8..15 -> k = kt*32+16+hi*8+(s-8). Both runs are 8 contiguous K
// values and never cross a 32-aligned source boundary -> float4 loads +
// v_cvt_pkrtz packs (branchless).
template <typename F>
static __device__ __forceinline__ v16h buildFrag(int kt, int lane, F src){
  int m = lane & 15, hi = lane >> 4;
  const float* r0 = src(m, kt*32 + hi*8);
  const float* r1 = src(m, kt*32 + 16 + hi*8);
  float4 x0 = *(const float4*)r0;
  float4 x1 = *(const float4*)(r0 + 4);
  float4 y0 = *(const float4*)r1;
  float4 y1 = *(const float4*)(r1 + 4);
  AFrag A;
  A.p[0] = __builtin_amdgcn_cvt_pkrtz(x0.x, x0.y);
  A.p[1] = __builtin_amdgcn_cvt_pkrtz(x0.z, x0.w);
  A.p[2] = __builtin_amdgcn_cvt_pkrtz(x1.x, x1.y);
  A.p[3] = __builtin_amdgcn_cvt_pkrtz(x1.z, x1.w);
  A.p[4] = __builtin_amdgcn_cvt_pkrtz(y0.x, y0.y);
  A.p[5] = __builtin_amdgcn_cvt_pkrtz(y0.z, y0.w);
  A.p[6] = __builtin_amdgcn_cvt_pkrtz(y1.x, y1.y);
  A.p[7] = __builtin_amdgcn_cvt_pkrtz(y1.z, y1.w);
  return A.v;
}

template <typename F>
static __device__ __forceinline__ void packFrag(f16* dstBase, int kt, int lane, F src){
  v16h a = buildFrag(kt, lane, src);
  *(v16h*)(dstBase + (long)kt*512 + lane*16) = a;
}

// ---------------------------------------------------------------------------
// Weight packing: B-operand tile (n0,k0): lane l holds n = n0+(l&15),
// k = k0 + (slot>>3)*16 + (l>>4)*8 + (slot&7) (matches A-fragment order).
// ---------------------------------------------------------------------------
__global__ void k_pack2(const float* __restrict__ W1, int K1,
                        const float* __restrict__ W2, int K2,
                        int Kt, f16* __restrict__ dst, long total){
  for (long idx = (long)blockIdx.x*blockDim.x + threadIdx.x; idx < total;
       idx += (long)gridDim.x*blockDim.x){
    int  r       = (int)(idx & 511);
    long tileIdx = idx >> 9;
    int  lane = r >> 4, slot = r & 15;
    int  nt = (int)(tileIdx / Kt), kt = (int)(tileIdx % Kt);
    int  n = nt*16 + (lane & 15);
    int  k = kt*32 + ((slot >> 3) << 4) + ((lane >> 4) << 3) + (slot & 7);
    float v;
    if (k < K1)      v = W1[(long)n*K1 + k];
    else if (W2)     v = W2[(long)n*K2 + (k - K1)];
    else             v = 0.f;
    dst[idx] = (f16)v;
  }
}

// proj(80)+gate(1) packed as one N=96 (zero-padded) matrix over K=1536=[dh|ctx]
__global__ void k_packPG(const float* __restrict__ Wproj, const float* __restrict__ Wgate,
                         f16* __restrict__ dst){
  long total = 6L*48*512;
  for (long idx = (long)blockIdx.x*blockDim.x + threadIdx.x; idx < total;
       idx += (long)gridDim.x*blockDim.x){
    int  r = (int)(idx & 511);
    long tileIdx = idx >> 9;
    int  lane = r >> 4, slot = r & 15;
    int  nt = (int)(tileIdx / 48), kt = (int)(tileIdx % 48);
    int  n = nt*16 + (lane & 15);
    int  k = kt*32 + ((slot >> 3) << 4) + ((lane >> 4) << 3) + (slot & 7);
    float v = 0.f;
    if (n < 80)       v = Wproj[(long)n*1536 + k];
    else if (n == 80) v = Wgate[k];
    dst[idx] = (f16)v;
  }
}

__global__ void k_zero(float* p, int n){
  int i = blockIdx.x*blockDim.x + threadIdx.x;
  if (i < n) p[i] = 0.f;
}

// prenet stage 1: h1 = relu(di @ Wpre1^T), di row t = (t==0 ? 0 : dec[:,:,t-1])
// K=80 padded to 96. For unrolled kt: kt<2 runs all <80; kt==2 run A (64..79)
// valid, run B (80..95) compile-time zero -> fully branchless.
__global__ void __launch_bounds__(512) k_prenet1(const float* __restrict__ dec,
                                                 const f16* __restrict__ Wp1,
                                                 float* __restrict__ h1){
  int t = blockIdx.x;                       // 0..399 : one M-tile (16 batch rows)
  int lane = threadIdx.x & 31, wv = threadIdx.x >> 5;   // 16 waves = 16 N-tiles
  int m = lane & 15, hi = lane >> 4;
  v8f acc = {0.f,0.f,0.f,0.f,0.f,0.f,0.f,0.f};
  if (t > 0){
    const float* base = dec + (long)m*NMEL*T_ + (t-1);
    #pragma unroll
    for (int kt = 0; kt < 3; ++kt){
      AFrag A;
      int ka = kt*32 + hi*8;
      int kb = kt*32 + 16 + hi*8;
      #pragma unroll
      for (int j = 0; j < 4; ++j)
        A.p[j] = __builtin_amdgcn_cvt_pkrtz(base[(long)(ka+2*j)*T_], base[(long)(ka+2*j+1)*T_]);
      #pragma unroll
      for (int j = 0; j < 4; ++j)
        A.p[4+j] = (kt == 2) ? __builtin_amdgcn_cvt_pkrtz(0.f, 0.f)
                             : __builtin_amdgcn_cvt_pkrtz(base[(long)(kb+2*j)*T_], base[(long)(kb+2*j+1)*T_]);
      v16h b = *(const v16h*)(Wp1 + ((long)wv*3 + kt)*512 + lane*16);
      acc = wmma16(A.v, b, acc);
    }
  }
  int n = wv*16 + (lane & 15);
  int mb = (lane >> 4)*8;
  #pragma unroll
  for (int r = 0; r < 8; ++r)
    h1[((long)t*16 + mb + r)*PRE_ + n] = fmaxf(acc[r], 0.f);
}

// prenet stage 2: xs = relu(h1 @ Wpre2^T)
__global__ void __launch_bounds__(512) k_prenet2(const float* __restrict__ h1,
                                                 const f16* __restrict__ Wp2,
                                                 float* __restrict__ xs){
  int t = blockIdx.x;
  int lane = threadIdx.x & 31, wv = threadIdx.x >> 5;
  v8f acc = {0.f,0.f,0.f,0.f,0.f,0.f,0.f,0.f};
  auto src = [&](int m, int k)->const float*{ return h1 + ((long)t*16 + m)*PRE_ + k; };
  #pragma unroll 2
  for (int kt = 0; kt < 8; ++kt){           // K = 256
    v16h a = buildFrag(kt, lane, src);
    v16h b = *(const v16h*)(Wp2 + ((long)wv*8 + kt)*512 + lane*16);
    acc = wmma16(a, b, acc);
  }
  int n = wv*16 + (lane & 15);
  int mb = (lane >> 4)*8;
  #pragma unroll
  for (int r = 0; r < 8; ++r)
    xs[((long)t*16 + mb + r)*PRE_ + n] = fmaxf(acc[r], 0.f);
}

// proc_mem = memory @ Wm^T  (one-time, fp32 VALU)
__global__ void k_procmem(const float* __restrict__ memory,
                          const float* __restrict__ Wm,
                          float* __restrict__ pm){
  int idx = blockIdx.x*blockDim.x + threadIdx.x;   // 4096*128 threads
  int a = idx & 127, row = idx >> 7;
  const float4* mr = (const float4*)(memory + (long)row*ENC_);
  const float4* wr = (const float4*)(Wm + (long)a*ENC_);
  float acc = 0.f;
  #pragma unroll 4
  for (int k = 0; k < ENC_/4; ++k){
    float4 x = mr[k], w = wr[k];
    acc += x.x*w.x + x.y*w.y + x.z*w.z + x.w*w.w;
  }
  pm[idx] = acc;
}

// ---------------------------------------------------------------------------
// Persistent decoder kernel
// ---------------------------------------------------------------------------
struct DecP {
  const float *memory, *Wq, *bih_a, *bhh_a, *bih_d, *bhh_d;
  const float *Wconv, *Wld, *vvec, *bproj, *bgate;
  const int*  memlen;
  const f16  *WaP, *WdP, *WpgP;
  f16 *XaP, *XdP, *XpP;
  const float *xs, *pm;
  float *Za, *Zd, *Zp;
  float *ah, *ac, *dh, *dc, *aw, *awc, *ctx;
  float *outMel, *outGate, *outAlign;
  int* bar;
};

static __device__ void gsync(int* bar){
  __threadfence();
  __syncthreads();
  if (threadIdx.x == 0){
    int gen  = __hip_atomic_load(bar+1, __ATOMIC_ACQUIRE, __HIP_MEMORY_SCOPE_AGENT);
    int prev = __hip_atomic_fetch_add(bar, 1, __ATOMIC_ACQ_REL, __HIP_MEMORY_SCOPE_AGENT);
    if (prev == NBLK-1){
      __hip_atomic_store(bar, 0, __ATOMIC_RELAXED, __HIP_MEMORY_SCOPE_AGENT);
      __hip_atomic_fetch_add(bar+1, 1, __ATOMIC_RELEASE, __HIP_MEMORY_SCOPE_AGENT);
    } else {
      while (__hip_atomic_load(bar+1, __ATOMIC_ACQUIRE, __HIP_MEMORY_SCOPE_AGENT) == gen)
        __builtin_amdgcn_s_sleep(1);
    }
  }
  __syncthreads();
}

__global__ void __launch_bounds__(THREADS) k_decoder(DecP p){
  const int tid  = threadIdx.x;
  const int lane = tid & 31;
  const int gw   = blockIdx.x * WPB + (tid >> 5);    // 0..511
  const int gt   = blockIdx.x * THREADS + tid;       // 0..16383

  __shared__ float sWc[NF_*2*KC_];    // 1984: location conv weights
  __shared__ float sWld[ATT_*NF_];    // 4096
  __shared__ float sv[ATT_];
  __shared__ float sAw[S_], sAwc[S_], sAwN[S_];
  __shared__ float sQ[ATT_], sRed[S_];
  for (int i = tid; i < NF_*2*KC_; i += THREADS) sWc[i]  = p.Wconv[i];
  for (int i = tid; i < ATT_*NF_;  i += THREADS) sWld[i] = p.Wld[i];
  for (int i = tid; i < ATT_;      i += THREADS) sv[i]   = p.vvec[i];
  __syncthreads();

  for (int t = 0; t < T_; ++t){
    // ---- S1: emit step t-1 mel/gate  +  pack Xa = [x|ctx|ah] (56 kt)
    if (t > 0 && gt < 16*81){
      int b = gt/81, n = gt - b*81;
      float z = p.Zp[b*96 + n];
      if (n < 80) p.outMel[((long)b*80 + n)*T_ + (t-1)] = z + p.bproj[n];
      else        p.outGate[(long)b*T_ + (t-1)]         = z + p.bgate[0];
    }
    if (gw < 56){
      auto src = [&](int m, int k)->const float*{
        if (k < 256) return p.xs + ((long)t*16 + m)*256 + k;
        if (k < 768) return p.ctx + m*512 + (k-256);
        return p.ah + m*1024 + (k-768);
      };
      packFrag(p.XaP, gw, lane, src);
    }
    gsync(p.bar);

    // ---- S2: Za = Xa @ Wa^T   (256 N-tiles x 2 K-halves = 512 wave slots)
    {
      int tile = gw >> 1, half = gw & 1;
      int kb = half ? 28 : 0;
      v8f acc = {0.f,0.f,0.f,0.f,0.f,0.f,0.f,0.f};
      acc = gemm_loop<28>(p.XaP + (long)kb*512 + lane*16,
                          p.WaP + ((long)tile*56 + kb)*512 + lane*16, acc);
      float* Z = p.Za + (long)half*65536;
      int n = tile*16 + (lane & 15);
      int mb = (lane >> 4)*8;
      #pragma unroll
      for (int r = 0; r < 8; ++r) Z[(long)(mb + r)*4096 + n] = acc[r];
    }
    gsync(p.bar);

    // ---- S3: attention LSTM cell (16x1024 = exactly one elem/thread)
    {
      int b = gt >> 10, u = gt & 1023;
      const float* z0 = p.Za + (long)b*4096;
      const float* z1 = z0 + 65536;
      float zi = z0[u]      + z1[u]      + p.bih_a[u]      + p.bhh_a[u];
      float zf = z0[u+1024] + z1[u+1024] + p.bih_a[u+1024] + p.bhh_a[u+1024];
      float zg = z0[u+2048] + z1[u+2048] + p.bih_a[u+2048] + p.bhh_a[u+2048];
      float zo = z0[u+3072] + z1[u+3072] + p.bih_a[u+3072] + p.bhh_a[u+3072];
      float cn = sigm(zf)*p.ac[gt] + sigm(zi)*tanhf(zg);
      p.ac[gt] = cn;
      p.ah[gt] = sigm(zo)*tanhf(cn);
    }
    gsync(p.bar);

    // ---- S4: block-local attention megastage (blocks 0..15 own batch row b):
    //          q -> location conv -> energies -> softmax -> aw/awc/align -> ctx
    if (blockIdx.x < 16){
      int b = blockIdx.x, s = tid;
      sAw[tid]  = p.aw[b*256 + tid];
      sAwc[tid] = p.awc[b*256 + tid];
      // q = ah[b] @ Wq^T : 128 outs, 2 threads per out (512 MACs each)
      {
        int a = tid >> 1, hf = tid & 1;
        const float4* hr = (const float4*)(p.ah + (long)b*1024 + hf*512);
        const float4* wr = (const float4*)(p.Wq + (long)a*1024 + hf*512);
        float acc = 0.f;
        #pragma unroll 4
        for (int j = 0; j < 128; ++j){
          float4 x = hr[j], w = wr[j];
          acc += x.x*w.x + x.y*w.y + x.z*w.z + x.w*w.w;
        }
        sRed[tid] = acc;
      }
      __syncthreads();
      if (tid < ATT_) sQ[tid] = sRed[2*tid] + sRed[2*tid+1];
      __syncthreads();
      // location conv (registers) + energies
      float loc[NF_];
      for (int f = 0; f < NF_; ++f){
        float acc = 0.f;
        for (int k = 0; k < KC_; ++k){
          int sp = s + k - 15;
          if (sp >= 0 && sp < 256)
            acc += sAw[sp]*sWc[f*62 + k] + sAwc[sp]*sWc[f*62 + 31 + k];
        }
        loc[f] = acc;
      }
      const float* pmr = p.pm + ((long)b*256 + s)*128;
      float ebs = 0.f;
      for (int a = 0; a < ATT_; ++a){
        float tv = sQ[a] + pmr[a];
        #pragma unroll
        for (int f = 0; f < NF_; ++f) tv += loc[f]*sWld[a*32 + f];
        ebs += tanhf(tv)*sv[a];
      }
      if (s >= p.memlen[b]) ebs = -1.0e9f;
      // block softmax over S=256
      sRed[tid] = ebs; __syncthreads();
      for (int o = 128; o > 0; o >>= 1){
        if (tid < o) sRed[tid] = fmaxf(sRed[tid], sRed[tid+o]);
        __syncthreads();
      }
      float vmax = sRed[0]; __syncthreads();
      float ex = __expf(ebs - vmax);
      sRed[tid] = ex; __syncthreads();
      for (int o = 128; o > 0; o >>= 1){
        if (tid < o) sRed[tid] += sRed[tid+o];
        __syncthreads();
      }
      float pr = ex / sRed[0];
      p.aw[b*256 + s]  = pr;
      p.awc[b*256 + s] = sAwc[s] + pr;
      p.outAlign[(long)b*T_*S_ + (long)t*S_ + s] = pr;
      sAwN[tid] = pr;
      __syncthreads();
      // ctx = aw_new @ memory[b] : 512 outs, 2 per thread (coalesced in e)
      #pragma unroll
      for (int eo = 0; eo < 2; ++eo){
        int ei = tid + eo*256;
        const float* mr = p.memory + (long)b*256*512 + ei;
        float acc = 0.f;
        for (int s2 = 0; s2 < 256; ++s2) acc += sAwN[s2]*mr[(long)s2*512];
        p.ctx[b*512 + ei] = acc;
      }
    }
    gsync(p.bar);

    // ---- S5: pack Xd = [ah|ctx|dh] (80 kt)
    if (gw < 80){
      auto src = [&](int m, int k)->const float*{
        if (k < 1024) return p.ah + m*1024 + k;
        if (k < 1536) return p.ctx + m*512 + (k-1024);
        return p.dh + m*1024 + (k-1536);
      };
      packFrag(p.XdP, gw, lane, src);
    }
    gsync(p.bar);

    // ---- S6: Zd = Xd @ Wd^T   (256 tiles x 2 K-halves)
    {
      int tile = gw >> 1, half = gw & 1;
      int kb = half ? 40 : 0;
      v8f acc = {0.f,0.f,0.f,0.f,0.f,0.f,0.f,0.f};
      acc = gemm_loop<40>(p.XdP + (long)kb*512 + lane*16,
                          p.WdP + ((long)tile*80 + kb)*512 + lane*16, acc);
      float* Z = p.Zd + (long)half*65536;
      int n = tile*16 + (lane & 15);
      int mb = (lane >> 4)*8;
      #pragma unroll
      for (int r = 0; r < 8; ++r) Z[(long)(mb + r)*4096 + n] = acc[r];
    }
    gsync(p.bar);

    // ---- S7: decoder LSTM cell
    {
      int b = gt >> 10, u = gt & 1023;
      const float* z0 = p.Zd + (long)b*4096;
      const float* z1 = z0 + 65536;
      float zi = z0[u]      + z1[u]      + p.bih_d[u]      + p.bhh_d[u];
      float zf = z0[u+1024] + z1[u+1024] + p.bih_d[u+1024] + p.bhh_d[u+1024];
      float zg = z0[u+2048] + z1[u+2048] + p.bih_d[u+2048] + p.bhh_d[u+2048];
      float zo = z0[u+3072] + z1[u+3072] + p.bih_d[u+3072] + p.bhh_d[u+3072];
      float cn = sigm(zf)*p.dc[gt] + sigm(zi)*tanhf(zg);
      p.dc[gt] = cn;
      p.dh[gt] = sigm(zo)*tanhf(cn);
    }
    gsync(p.bar);

    // ---- S8: pack Xp = [dh|ctx] (48 kt)
    if (gw < 48){
      auto src = [&](int m, int k)->const float*{
        return (k < 1024) ? p.dh + m*1024 + k : p.ctx + m*512 + (k-1024);
      };
      packFrag(p.XpP, gw, lane, src);
    }
    gsync(p.bar);

    // ---- S9: Zp = Xp @ [Wproj;Wgate]^T  (6 N-tiles, K=48 kt)
    if (gw < 6){
      v8f acc = {0.f,0.f,0.f,0.f,0.f,0.f,0.f,0.f};
      acc = gemm_loop<48>(p.XpP + lane*16,
                          p.WpgP + (long)gw*48*512 + lane*16, acc);
      int n = gw*16 + (lane & 15);
      int mb = (lane >> 4)*8;
      #pragma unroll
      for (int r = 0; r < 8; ++r) p.Zp[(long)(mb + r)*96 + n] = acc[r];
    }
    gsync(p.bar);
  }

  // tail: emit final step's mel/gate (Zp visible after last gsync)
  if (gt < 16*81){
    int b = gt/81, n = gt - b*81;
    float z = p.Zp[b*96 + n];
    if (n < 80) p.outMel[((long)b*80 + n)*T_ + (T_-1)] = z + p.bproj[n];
    else        p.outGate[(long)b*T_ + (T_-1)]         = z + p.bgate[0];
  }
}

// ---------------------------------------------------------------------------
extern "C" void kernel_launch(void* const* d_in, const int* in_sizes, int n_in,
                              void* d_out, int out_size, void* d_ws, size_t ws_size,
                              hipStream_t stream) {
  (void)in_sizes; (void)n_in; (void)out_size; (void)ws_size;
  const float* memory = (const float*)d_in[0];
  const float* dec    = (const float*)d_in[1];
  const float* Wpre1  = (const float*)d_in[2];
  const float* Wpre2  = (const float*)d_in[3];
  const float* Wih_a  = (const float*)d_in[4];
  const float* Whh_a  = (const float*)d_in[5];
  const float* bih_a  = (const float*)d_in[6];
  const float* bhh_a  = (const float*)d_in[7];
  const float* Wq     = (const float*)d_in[8];
  const float* Wm     = (const float*)d_in[9];
  const float* vvec   = (const float*)d_in[10];
  const float* Wconv  = (const float*)d_in[11];
  const float* Wld    = (const float*)d_in[12];
  const float* Wih_d  = (const float*)d_in[13];
  const float* Whh_d  = (const float*)d_in[14];
  const float* bih_d  = (const float*)d_in[15];
  const float* bhh_d  = (const float*)d_in[16];
  const float* Wproj  = (const float*)d_in[17];
  const float* bproj  = (const float*)d_in[18];
  const float* Wgate  = (const float*)d_in[19];
  const float* bgate  = (const float*)d_in[20];
  const int*   memlen = (const int*)d_in[21];

  char* ws = (char*)d_ws;
  size_t off = 0;
  auto take = [&](size_t bytes)->char*{
    char* r = ws + off;
    off += bytes;
    off = (off + 255) & ~(size_t)255;
    return r;
  };
  f16*   WaP  = (f16*)take(7340032UL*2);   // 256 x 56 tiles
  f16*   WdP  = (f16*)take(10485760UL*2);  // 256 x 80 tiles
  f16*   WpgP = (f16*)take(147456UL*2);    // 6 x 48 tiles
  f16*   Wp1P = (f16*)take(24576UL*2);     // 16 x 3 tiles
  f16*   Wp2P = (f16*)take(65536UL*2);     // 16 x 8 tiles
  f16*   XaP  = (f16*)take(28672UL*2);
  f16*   XdP  = (f16*)take(40960UL*2);
  f16*   XpP  = (f16*)take(24576UL*2);
  float* xs   = (float*)take(1638400UL*4);
  float* h1   = (float*)take(1638400UL*4);
  float* pm   = (float*)take(524288UL*4);
  float* Za   = (float*)take(131072UL*4);  // two K-halves
  float* Zd   = (float*)take(131072UL*4);
  float* Zp   = (float*)take(1536UL*4);
  float* st   = (float*)take(81936UL*4);   // state block (zeroed each call)

  float* ah  = st;
  float* ac  = st + 16384;
  float* dh  = st + 32768;
  float* dc  = st + 49152;
  float* aw  = st + 65536;
  float* awc = st + 69632;
  float* ctx = st + 73728;
  int*   bar = (int*)(st + 81920);

  float* outMel   = (float*)d_out;
  float* outGate  = outMel + (long)B_*NMEL*T_;     // 512000
  float* outAlign = outGate + (long)B_*T_;         // +6400

  // zero state + barrier
  k_zero<<<(81936 + 255)/256, 256, 0, stream>>>(st, 81936);

  // pack weights (f16 WMMA B-fragment layout)
  k_pack2<<<4096, 256, 0, stream>>>(Wih_a, 768,  Whh_a, 1024, 56, WaP, 7340032L);
  k_pack2<<<4096, 256, 0, stream>>>(Wih_d, 1536, Whh_d, 1024, 80, WdP, 10485760L);
  k_pack2<<<128,  256, 0, stream>>>(Wpre1, 80,   (const float*)nullptr, 0, 3, Wp1P, 24576L);
  k_pack2<<<256,  256, 0, stream>>>(Wpre2, 256,  (const float*)nullptr, 0, 8, Wp2P, 65536L);
  k_packPG<<<576, 256, 0, stream>>>(Wproj, Wgate, WpgP);

  // parallel pre-work
  k_prenet1<<<T_, 512, 0, stream>>>(dec, Wp1P, h1);
  k_prenet2<<<T_, 512, 0, stream>>>(h1, Wp2P, xs);
  k_procmem<<<2048, 256, 0, stream>>>(memory, Wm, pm);

  // persistent sequential decoder
  DecP P;
  P.memory = memory; P.Wq = Wq;
  P.bih_a = bih_a; P.bhh_a = bhh_a; P.bih_d = bih_d; P.bhh_d = bhh_d;
  P.Wconv = Wconv; P.Wld = Wld; P.vvec = vvec; P.bproj = bproj; P.bgate = bgate;
  P.memlen = memlen;
  P.WaP = WaP; P.WdP = WdP; P.WpgP = WpgP;
  P.XaP = XaP; P.XdP = XdP; P.XpP = XpP;
  P.xs = xs; P.pm = pm;
  P.Za = Za; P.Zd = Zd; P.Zp = Zp;
  P.ah = ah; P.ac = ac; P.dh = dh; P.dc = dc;
  P.aw = aw; P.awc = awc; P.ctx = ctx;
  P.outMel = outMel; P.outGate = outGate; P.outAlign = outAlign;
  P.bar = bar;
  k_decoder<<<NBLK, THREADS, 0, stream>>>(P);
}
